// LatentBridgeVQ_67963562491993
// MI455X (gfx1250) — matrix operand order repaired and checked by reading, not compiled, not run on hardware
//
#include <hip/hip_runtime.h>
#include <hip/hip_bf16.h>
#include <type_traits>

// ---------------- CDNA5 / gfx1250 types ----------------
typedef __bf16 bf16;
typedef __attribute__((ext_vector_type(16))) __bf16 v16bf;
typedef __attribute__((ext_vector_type(8)))  __bf16 bf16x8;
typedef __attribute__((ext_vector_type(8)))  float  v8f;

// ---------------- problem constants ----------------
#define BB     8
#define TT     2048
#define SRCD   4096
#define ED     2048          // TGT
#define KL     32            // latents
#define HH     8
#define HD     256
#define DEPTH  2
#define CODES  4096
#define MKEYS  (BB*TT)       // 16384
#define MX     (BB*KL)       // 256
#define OUTN   (MX*ED)       // 524288

#define LDSK   40            // LDS row pitch (bf16 elems) for 32-K tiles; 80B rows keep 16B alignment

// =====================================================================
//  WMMA GEMM:  C[M,N] = act( A[M,Kd] * B[N,Kd]^T + bias )  (+ residual)
//  A: float or bf16 row-major, B: float row-major [N,Kd] (i.e. x @ W^T)
//  Block tile 128x128, 8 waves, each wave 32x64 (2x4 accumulators).
// =====================================================================
template<typename AT, typename OT, bool RESID, bool GELU>
__global__ __launch_bounds__(256)
void gemm_xwT(const AT* __restrict__ A, const float* __restrict__ Bw,
              const float* __restrict__ bias, OT* __restrict__ C,
              int M, int N, int Kd)
{
    __shared__ bf16 lsA[128 * LDSK];
    __shared__ bf16 lsB[128 * LDSK];

    const int tid   = threadIdx.x;
    const int lane  = tid & 31;
    const int wid   = tid >> 5;         // 0..7
    const int waveM = wid >> 1;         // 0..3  -> 32 rows each
    const int waveN = wid & 1;          // 0..1  -> 64 cols each
    const int l16   = lane & 15;
    const int lhalf = lane >> 4;

    const int mBase = blockIdx.y * 128;
    const int nBase = blockIdx.x * 128;

    v8f acc[2][4];
#pragma unroll
    for (int i = 0; i < 2; ++i)
#pragma unroll
        for (int t = 0; t < 4; ++t)
#pragma unroll
            for (int e = 0; e < 8; ++e) acc[i][t][e] = 0.0f;

    for (int k0 = 0; k0 < Kd; k0 += 32) {
        __syncthreads();
        // ---- load A tile (128x32) -> bf16 LDS ----
        if constexpr (std::is_same<AT, float>::value) {
#pragma unroll
            for (int it = 0; it < 4; ++it) {
                int s   = tid + it * 256;          // 1024 float4 slots
                int row = s >> 3;
                int c4  = (s & 7) * 4;
                const float* gp = A + (size_t)(mBase + row) * Kd + k0 + c4;
                if (k0 + 32 < Kd) __builtin_prefetch(gp + 32, 0, 3);
                float4 f = *(const float4*)gp;
                bf16* lp = &lsA[row * LDSK + c4];
                lp[0] = (bf16)f.x; lp[1] = (bf16)f.y;
                lp[2] = (bf16)f.z; lp[3] = (bf16)f.w;
            }
        } else {
#pragma unroll
            for (int it = 0; it < 2; ++it) {
                int s   = tid + it * 256;          // 512 bf16x8 slots
                int row = s >> 2;
                int c8  = (s & 3) * 8;
                const bf16* gp = A + (size_t)(mBase + row) * Kd + k0 + c8;
                if (k0 + 32 < Kd) __builtin_prefetch(gp + 32, 0, 3);
                *(bf16x8*)(&lsA[row * LDSK + c8]) = *(const bf16x8*)gp;
            }
        }
        // ---- load B tile (128 N-rows x 32 K) -> bf16 LDS ----
#pragma unroll
        for (int it = 0; it < 4; ++it) {
            int s   = tid + it * 256;
            int row = s >> 3;
            int c4  = (s & 7) * 4;
            const float* gp = Bw + (size_t)(nBase + row) * Kd + k0 + c4;
            if (k0 + 32 < Kd) __builtin_prefetch(gp + 32, 0, 3);
            float4 f = *(const float4*)gp;
            bf16* lp = &lsB[row * LDSK + c4];
            lp[0] = (bf16)f.x; lp[1] = (bf16)f.y;
            lp[2] = (bf16)f.z; lp[3] = (bf16)f.w;
        }
        __syncthreads();

        // ---- fragments per ISA 7.12.2 layouts ----
        v16bf afrag[2];
#pragma unroll
        for (int i = 0; i < 2; ++i) {
            int mrow = waveM * 32 + i * 16 + l16;
            const bf16* pr = &lsA[mrow * LDSK + lhalf * 8];
            bf16x8 lo = *(const bf16x8*)pr;         // K = kbase..kbase+7
            bf16x8 hi = *(const bf16x8*)(pr + 16);  // K = kbase+16..kbase+23
            afrag[i] = __builtin_shufflevector(lo, hi,
                0,1,2,3,4,5,6,7,8,9,10,11,12,13,14,15);
        }
        v16bf bfrag[4];
#pragma unroll
        for (int t = 0; t < 4; ++t) {
            int nrow = waveN * 64 + t * 16 + l16;
            const bf16* pr = &lsB[nrow * LDSK + lhalf * 8];  // K = half*8 + e, then 16 + half*8 + e
            bf16x8 lo = *(const bf16x8*)pr;
            bf16x8 hi = *(const bf16x8*)(pr + 16);
            bfrag[t] = __builtin_shufflevector(lo, hi,
                0,1,2,3,4,5,6,7,8,9,10,11,12,13,14,15);
        }
#pragma unroll
        for (int i = 0; i < 2; ++i)
#pragma unroll
            for (int t = 0; t < 4; ++t)
                acc[i][t] = __builtin_amdgcn_wmma_f32_16x16x32_bf16(
                    false, afrag[i], false, bfrag[t],
                    (short)0, acc[i][t], false, false);
    }

    // ---- epilogue: D layout -> row M=(lane>>4)*8+j, col N=lane&15 ----
#pragma unroll
    for (int i = 0; i < 2; ++i) {
#pragma unroll
        for (int t = 0; t < 4; ++t) {
            int cc = nBase + waveN * 64 + t * 16 + l16;
            float bv = bias ? bias[cc] : 0.0f;
#pragma unroll
            for (int j = 0; j < 8; ++j) {
                int r = mBase + waveM * 32 + i * 16 + lhalf * 8 + j;
                float v = acc[i][t][j] + bv;
                if constexpr (GELU)
                    v = 0.5f * v * (1.0f + erff(v * 0.70710678118654752440f));
                size_t off = (size_t)r * N + cc;
                if constexpr (RESID) v += ((const float*)C)[off];
                C[off] = (OT)v;
            }
        }
    }
}

// =====================================================================
//  helpers
// =====================================================================
__device__ inline float blk_sum(float v, float* red) {
    int t = threadIdx.x; red[t] = v; __syncthreads();
    for (int s = 128; s > 0; s >>= 1) { if (t < s) red[t] += red[t + s]; __syncthreads(); }
    float r = red[0]; __syncthreads(); return r;
}
__device__ inline float blk_max(float v, float* red) {
    int t = threadIdx.x; red[t] = v; __syncthreads();
    for (int s = 128; s > 0; s >>= 1) { if (t < s) red[t] = fmaxf(red[t], red[t + s]); __syncthreads(); }
    float r = red[0]; __syncthreads(); return r;
}

// ---- broadcast latents into x : [B,K,E] ----
__global__ __launch_bounds__(256)
void init_x_kernel(const float* __restrict__ lat, float* __restrict__ x) {
    int j = blockIdx.x * 256 + threadIdx.x;       // 0 .. OUTN-1
    x[j] = lat[j & (KL * ED - 1)];                // KL*ED = 65536 (pow2)
}

// ---- LayerNorm over last dim (E=2048), one block per row ----
__global__ __launch_bounds__(256)
void ln_kernel(const float* __restrict__ x, const float* __restrict__ g,
               const float* __restrict__ b, float* __restrict__ y) {
    __shared__ float red[256];
    int row = blockIdx.x, tid = threadIdx.x;
    const float* xr = x + (size_t)row * ED;
    float s = 0.f, s2 = 0.f;
    for (int i = tid; i < ED; i += 256) { float v = xr[i]; s += v; s2 += v * v; }
    float S  = blk_sum(s,  red);
    float S2 = blk_sum(s2, red);
    float mu  = S * (1.0f / ED);
    float var = S2 * (1.0f / ED) - mu * mu;
    float rs  = rsqrtf(var + 1e-5f);
    float* yr = y + (size_t)row * ED;
    for (int i = tid; i < ED; i += 256)
        yr[i] = (xr[i] - mu) * rs * g[i] + b[i];
}

// ---- attention: one block per (b,h,qi); q f32, k/v bf16 ----
__global__ __launch_bounds__(256)
void attention_kernel(const float* __restrict__ q, const bf16* __restrict__ k,
                      const bf16* __restrict__ v, const unsigned char* __restrict__ mask,
                      float* __restrict__ o, int Lq, int Lk) {
    __shared__ float sc[2048];
    __shared__ float qv[HD];
    __shared__ float red[256];
    int tid = threadIdx.x;
    int b  = blockIdx.x / (HH * Lq);
    int h  = (blockIdx.x / Lq) % HH;
    int qi = blockIdx.x % Lq;

    qv[tid] = q[(size_t)(b * Lq + qi) * ED + h * HD + tid];
    __syncthreads();

    const float scale = 0.0625f;  // 1/sqrt(256)
    float mx = -1e30f;
    for (int kk = tid; kk < Lk; kk += 256) {
        const bf16* kr = k + (size_t)(b * Lk + kk) * ED + h * HD;
        float d = 0.f;
#pragma unroll
        for (int dd = 0; dd < HD; dd += 8) {
            bf16x8 kv = *(const bf16x8*)(kr + dd);
#pragma unroll
            for (int e = 0; e < 8; ++e) d += qv[dd + e] * (float)kv[e];
        }
        d *= scale;
        if (mask && mask[(size_t)b * Lk + kk] == 0) d += -1e9f;
        sc[kk] = d;
        mx = fmaxf(mx, d);
    }
    float M = blk_max(mx, red);
    float sum = 0.f;
    for (int kk = tid; kk < Lk; kk += 256) {
        float e = __expf(sc[kk] - M);
        sc[kk] = e; sum += e;
    }
    float S = blk_sum(sum, red);
    float inv = 1.0f / S;
    __syncthreads();

    // one thread per head-dim
    float acc = 0.f;
    const bf16* vb = v + (size_t)b * Lk * ED + h * HD + tid;
    for (int kk = 0; kk < Lk; ++kk)
        acc += sc[kk] * (float)vb[(size_t)kk * ED];
    o[(size_t)(b * Lq + qi) * ED + h * HD + tid] = acc * inv;
}

// ---- codebook row norms ----
__global__ __launch_bounds__(256)
void cbnorm_kernel(const float* __restrict__ cb, float* __restrict__ cbn) {
    __shared__ float red[256];
    int n = blockIdx.x, tid = threadIdx.x;
    const float* r = cb + (size_t)n * ED;
    float s = 0.f;
    for (int i = tid; i < ED; i += 256) { float v = r[i]; s += v * v; }
    float S = blk_sum(s, red);
    if (tid == 0) cbn[n] = S;
}

// ---- per-row argmin over codes: d = ||x||^2 + ||c||^2 - 2 x.c ----
__global__ __launch_bounds__(256)
void argmin_kernel(const float* __restrict__ x, const float* __restrict__ dots,
                   const float* __restrict__ cbn, int* __restrict__ idx,
                   float* __restrict__ dist) {
    __shared__ float red[256];
    __shared__ float rv[256];
    __shared__ int   ri[256];
    int m = blockIdx.x, tid = threadIdx.x;
    const float* xr = x + (size_t)m * ED;
    float s = 0.f;
    for (int i = tid; i < ED; i += 256) { float v = xr[i]; s += v * v; }
    float rown = blk_sum(s, red);

    float bv = 1e30f; int bi = 0x7fffffff;
    const float* dr = dots + (size_t)m * CODES;
    for (int n = tid; n < CODES; n += 256) {
        float v = cbn[n] - 2.0f * dr[n];
        if (v < bv || (v == bv && n < bi)) { bv = v; bi = n; }
    }
    rv[tid] = bv; ri[tid] = bi; __syncthreads();
    for (int st = 128; st > 0; st >>= 1) {
        if (tid < st) {
            float ov = rv[tid + st]; int oi = ri[tid + st];
            if (ov < rv[tid] || (ov == rv[tid] && oi < ri[tid])) { rv[tid] = ov; ri[tid] = oi; }
        }
        __syncthreads();
    }
    if (tid == 0) { idx[m] = ri[0]; dist[m] = rown + rv[0]; }
}

// ---- losses + perplexity (single block) ----
__global__ __launch_bounds__(256)
void finalize_kernel(const int* __restrict__ idx, const float* __restrict__ dist,
                     float* __restrict__ out) {
    __shared__ int cnt[CODES];
    __shared__ float red[256];
    int tid = threadIdx.x;
    for (int i = tid; i < CODES; i += 256) cnt[i] = 0;
    __syncthreads();
    atomicAdd(&cnt[idx[tid]], 1);      // exactly MX=256 rows
    float dloc = dist[tid];
    __syncthreads();
    float ent = 0.f;
    for (int n = tid; n < CODES; n += 256) {
        float p = (float)cnt[n] * (1.0f / (float)MX);
        ent += -p * logf(p + 1e-10f);
    }
    float H  = blk_sum(ent,  red);
    float Ds = blk_sum(dloc, red);
    if (tid == 0) {
        out[OUTN]     = 1.25f * Ds / (float)(MX * ED);   // q_loss + 0.25*e_loss
        out[OUTN + 1] = expf(H);
    }
}

// ---- out = codebook[idx] * scale  (quant == qz numerically) ----
__global__ __launch_bounds__(256)
void out_kernel(const int* __restrict__ idx, const float* __restrict__ cb,
                const float* __restrict__ scale, float* __restrict__ out) {
    int j = blockIdx.x * 256 + threadIdx.x;   // 0..OUTN-1
    int m = j >> 11;                          // /ED
    int c = j & (ED - 1);
    out[j] = cb[(size_t)idx[m] * ED + c] * scale[0];
}

// =====================================================================
//  host-side launch
// =====================================================================
static inline size_t al256(size_t x) { return (x + 255) & ~(size_t)255; }

extern "C" void kernel_launch(void* const* d_in, const int* in_sizes, int n_in,
                              void* d_out, int out_size, void* d_ws, size_t ws_size,
                              hipStream_t stream) {
    (void)n_in; (void)out_size; (void)ws_size;
    // ---- input index mapping (handles insertion-order or tree-sorted) ----
    int pbase, isrc, imask;
    if (in_sizes[0] == BB * TT * SRCD) { isrc = 0; imask = 1; pbase = 2; }
    else                               { pbase = 0; isrc = 41; imask = 42; }

    const float* src   = (const float*)d_in[isrc];
    const unsigned char* mask = (const unsigned char*)d_in[imask];
    const float* cb    = (const float*)d_in[pbase + 0];
    const float* inp_b = (const float*)d_in[pbase + 1];
    const float* inp_w = (const float*)d_in[pbase + 2];
    const float* lat   = (const float*)d_in[pbase + 3];
    const float* scale = (const float*)d_in[pbase + 40];
    // layer fields (alphabetical within each layer dict)
    auto LP = [&](int l, int f) -> const float* {
        return (const float*)d_in[pbase + 4 + 18 * l + f];
    };
    enum { CA_IN_B=0, CA_IN_W, CA_OUT_B, CA_OUT_W, FFN_B1, FFN_B2, FFN_W1, FFN_W2,
           LN1_B, LN1_G, LN2_B, LN2_G, LN3_B, LN3_G, SA_IN_B, SA_IN_W, SA_OUT_B, SA_OUT_W };

    // ---- workspace carve ----
    char* w = (char*)d_ws; size_t off = 0;
    auto carve = [&](size_t bytes) { void* p = w + off; off += al256(bytes); return p; };
    bf16*  keys = (bf16*) carve((size_t)MKEYS * ED * 2);
    bf16*  kbuf = (bf16*) carve((size_t)MKEYS * ED * 2);
    bf16*  vbuf = (bf16*) carve((size_t)MKEYS * ED * 2);
    float* x    = (float*)carve((size_t)MX * ED * 4);
    float* xn   = (float*)carve((size_t)MX * ED * 4);
    float* qb   = (float*)carve((size_t)MX * ED * 4);
    float* ao   = (float*)carve((size_t)MX * ED * 4);
    float* ffnh = (float*)carve((size_t)MX * 4 * ED * 4);
    float* dots = (float*)carve((size_t)MX * CODES * 4);
    float* cbn  = (float*)carve((size_t)CODES * 4);
    float* dist = (float*)carve((size_t)MX * 4);
    int*   idxb = (int*)  carve((size_t)MX * 4);

    float* outp = (float*)d_out;
    dim3 blk(256);

    // ---- keys = src @ inp_w.T + inp_b  -> bf16 ----
    gemm_xwT<float, bf16, false, false>
        <<<dim3(ED / 128, MKEYS / 128), blk, 0, stream>>>(src, inp_w, inp_b, keys, MKEYS, ED, SRCD);

    // ---- x = broadcast(latents) ----
    init_x_kernel<<<OUTN / 256, blk, 0, stream>>>(lat, x);

    for (int l = 0; l < DEPTH; ++l) {
        // ===== cross attention =====
        ln_kernel<<<MX, blk, 0, stream>>>(x, LP(l, LN1_G), LP(l, LN1_B), xn);
        gemm_xwT<float, float, false, false>
            <<<dim3(ED / 128, MX / 128), blk, 0, stream>>>(xn, LP(l, CA_IN_W), LP(l, CA_IN_B), qb, MX, ED, ED);
        gemm_xwT<bf16, bf16, false, false>
            <<<dim3(ED / 128, MKEYS / 128), blk, 0, stream>>>(keys, LP(l, CA_IN_W) + (size_t)ED * ED,
                                                              LP(l, CA_IN_B) + ED, kbuf, MKEYS, ED, ED);
        gemm_xwT<bf16, bf16, false, false>
            <<<dim3(ED / 128, MKEYS / 128), blk, 0, stream>>>(keys, LP(l, CA_IN_W) + (size_t)2 * ED * ED,
                                                              LP(l, CA_IN_B) + 2 * ED, vbuf, MKEYS, ED, ED);
        attention_kernel<<<BB * HH * KL, blk, 0, stream>>>(qb, kbuf, vbuf, mask, ao, KL, TT);
        gemm_xwT<float, float, true, false>
            <<<dim3(ED / 128, MX / 128), blk, 0, stream>>>(ao, LP(l, CA_OUT_W), LP(l, CA_OUT_B), x, MX, ED, ED);

        // ===== self attention =====
        ln_kernel<<<MX, blk, 0, stream>>>(x, LP(l, LN2_G), LP(l, LN2_B), xn);
        gemm_xwT<float, float, false, false>
            <<<dim3(ED / 128, MX / 128), blk, 0, stream>>>(xn, LP(l, SA_IN_W), LP(l, SA_IN_B), qb, MX, ED, ED);
        gemm_xwT<float, bf16, false, false>
            <<<dim3(ED / 128, MX / 128), blk, 0, stream>>>(xn, LP(l, SA_IN_W) + (size_t)ED * ED,
                                                           LP(l, SA_IN_B) + ED, kbuf, MX, ED, ED);
        gemm_xwT<float, bf16, false, false>
            <<<dim3(ED / 128, MX / 128), blk, 0, stream>>>(xn, LP(l, SA_IN_W) + (size_t)2 * ED * ED,
                                                           LP(l, SA_IN_B) + 2 * ED, vbuf, MX, ED, ED);
        attention_kernel<<<BB * HH * KL, blk, 0, stream>>>(qb, kbuf, vbuf, nullptr, ao, KL, KL);
        gemm_xwT<float, float, true, false>
            <<<dim3(ED / 128, MX / 128), blk, 0, stream>>>(ao, LP(l, SA_OUT_W), LP(l, SA_OUT_B), x, MX, ED, ED);

        // ===== FFN =====
        ln_kernel<<<MX, blk, 0, stream>>>(x, LP(l, LN3_G), LP(l, LN3_B), xn);
        gemm_xwT<float, float, false, true>
            <<<dim3(4 * ED / 128, MX / 128), blk, 0, stream>>>(xn, LP(l, FFN_W1), LP(l, FFN_B1), ffnh, MX, 4 * ED, ED);
        gemm_xwT<float, float, true, false>
            <<<dim3(ED / 128, MX / 128), blk, 0, stream>>>(ffnh, LP(l, FFN_W2), LP(l, FFN_B2), x, MX, ED, 4 * ED);
    }

    // ===== vector quantize =====
    gemm_xwT<float, float, false, false>
        <<<dim3(CODES / 128, MX / 128), blk, 0, stream>>>(x, cb, nullptr, dots, MX, CODES, ED);
    cbnorm_kernel<<<CODES, blk, 0, stream>>>(cb, cbn);
    argmin_kernel<<<MX, blk, 0, stream>>>(x, dots, cbn, idxb, dist);
    finalize_kernel<<<1, blk, 0, stream>>>(idxb, dist, outp);
    out_kernel<<<OUTN / 256, blk, 0, stream>>>(idxb, cb, scale, outp);
}
